// MyModel_56968446214674
// MI455X (gfx1250) — matrix operand order repaired and checked by reading, not compiled, not run on hardware
//
#include <hip/hip_runtime.h>
#include <hip/hip_bf16.h>

typedef __attribute__((ext_vector_type(16))) _Float16 v16h;
typedef __attribute__((ext_vector_type(8)))  _Float16 v8h;
typedef __attribute__((ext_vector_type(8)))  float    v8f;

#define BN_EPS 1e-5f
#define NH 512          // hidden width
#define NF 4            // engineered features

// ---------------------------------------------------------------------------
// Kernel 0: zero the reduction accumulators (moments + BN2 sums)
// ---------------------------------------------------------------------------
__global__ void k0_zero(double* moments, float* sum2, float* sumsq2) {
    int t = threadIdx.x;
    if (t < 16) moments[t] = 0.0;
    for (int j = t; j < NH; j += blockDim.x) { sum2[j] = 0.f; sumsq2[j] = 0.f; }
}

// ---------------------------------------------------------------------------
// Bilinear interpolation (searchsorted-right minus 1, clamped)
// ---------------------------------------------------------------------------
__device__ __forceinline__ float inter2d(const float* xs, int nx,
                                         const float* ys, int ny,
                                         const float* tab,
                                         float xq, float yq) {
    xq = fminf(fmaxf(xq, xs[0]), xs[nx - 1]);
    yq = fminf(fmaxf(yq, ys[0]), ys[ny - 1]);
    int i = 0, j = 0;
    for (int t = 1; t <= nx - 2; ++t) if (xs[t] <= xq) i = t;
    for (int t = 1; t <= ny - 2; ++t) if (ys[t] <= yq) j = t;
    float x0 = xs[i], x1 = xs[i + 1];
    float y0 = ys[j], y1 = ys[j + 1];
    float tx = (xq - x0) / (x1 - x0);
    float ty = (yq - y0) / (y1 - y0);
    float f00 = tab[i * ny + j],       f01 = tab[i * ny + j + 1];
    float f10 = tab[(i + 1) * ny + j], f11 = tab[(i + 1) * ny + j + 1];
    return f00 * (1.f - tx) * (1.f - ty) + f10 * tx * (1.f - ty)
         + f01 * (1.f - tx) * ty        + f11 * tx * ty;
}

// ---------------------------------------------------------------------------
// Kernel 1: features + first/second moments of use_x1 (for analytic BN1 stats)
// ---------------------------------------------------------------------------
__global__ void __launch_bounds__(256)
k1_features(const float* __restrict__ x,
            const float* __restrict__ temp_set,
            const float* __restrict__ temp_envr,
            const float* __restrict__ tab,
            float* __restrict__ u4, double* __restrict__ moments, int B) {
    int i = blockIdx.x * 256 + threadIdx.x;
    float u[NF] = {0.f, 0.f, 0.f, 0.f};
    if (i < B) {
        const float* xr = x + (size_t)i * 15;
        float amb  = xr[1], sfl = xr[2], sfr = xr[3];
        float incar = xr[8], breq = xr[9], cool = xr[10];
        u[0] = inter2d(temp_set, 9, temp_envr, 14, tab, sfl, amb) - incar;
        u[1] = inter2d(temp_set, 9, temp_envr, 14, tab, sfr, amb) - incar;
        u[2] = amb;
        u[3] = breq - cool;
        float* dst = u4 + (size_t)i * NF;
        dst[0] = u[0]; dst[1] = u[1]; dst[2] = u[2]; dst[3] = u[3];
    }
    // 14 moment values: 4 means + 10 upper-tri second moments
    float mom[14];
    mom[0] = u[0]; mom[1] = u[1]; mom[2] = u[2]; mom[3] = u[3];
    int idx = 4;
    #pragma unroll
    for (int k = 0; k < NF; ++k)
        #pragma unroll
        for (int l = k; l < NF; ++l) mom[idx++] = u[k] * u[l];
    int lane = threadIdx.x & 31;
    #pragma unroll
    for (int m = 0; m < 14; ++m) {
        float v = mom[m];
        for (int o = 16; o > 0; o >>= 1) v += __shfl_xor(v, o, 32);
        if (lane == 0) atomicAdd(&moments[m], (double)v);
    }
}

// ---------------------------------------------------------------------------
// Kernel 2: analytic BN1 statistics.  h_pre1 = u @ w1 + b1 is linear in u, so
// mu = m.w + b, var = w^T Cov(u) w  from the 14 reduced moments.
// ---------------------------------------------------------------------------
__global__ void k2_bn1_stats(const double* __restrict__ moments,
                             const float* __restrict__ w1,
                             const float* __restrict__ b1,
                             float* __restrict__ mu1, float* __restrict__ inv1,
                             float invB) {
    int j = threadIdx.x;   // one block of 512
    float m[NF];
    #pragma unroll
    for (int k = 0; k < NF; ++k) m[k] = (float)(moments[k] * (double)invB);
    float c[NF][NF];
    int idx = 4;
    #pragma unroll
    for (int k = 0; k < NF; ++k)
        #pragma unroll
        for (int l = k; l < NF; ++l) {
            float cv = (float)(moments[idx++] * (double)invB) - m[k] * m[l];
            c[k][l] = cv; c[l][k] = cv;
        }
    float wv[NF];
    #pragma unroll
    for (int k = 0; k < NF; ++k) wv[k] = w1[k * NH + j];
    float mu = b1[j];
    #pragma unroll
    for (int k = 0; k < NF; ++k) mu += m[k] * wv[k];
    float var = 0.f;
    #pragma unroll
    for (int k = 0; k < NF; ++k)
        #pragma unroll
        for (int l = 0; l < NF; ++l) var += c[k][l] * wv[k] * wv[l];
    mu1[j]  = mu;
    inv1[j] = rsqrtf(var + BN_EPS);
}

// ---------------------------------------------------------------------------
// Kernel 3: h1 = tanh(BN1(u @ w1 + b1)) stored row-major f16 (WMMA A operand)
// ---------------------------------------------------------------------------
__global__ void __launch_bounds__(256)
k3_h1(const float* __restrict__ u4, const float* __restrict__ w1,
      const float* __restrict__ b1, const float* __restrict__ g1,
      const float* __restrict__ beta1, const float* __restrict__ mu1,
      const float* __restrict__ inv1, _Float16* __restrict__ h1) {
    size_t row = blockIdx.x;
    const float* u = u4 + row * NF;
    float u0 = u[0], u1 = u[1], u2 = u[2], u3 = u[3];
    for (int j = threadIdx.x; j < NH; j += 256) {
        float d = u0 * w1[j] + u1 * w1[NH + j] + u2 * w1[2 * NH + j]
                + u3 * w1[3 * NH + j] + b1[j];
        float h = tanhf(g1[j] * (d - mu1[j]) * inv1[j] + beta1[j]);
        h1[row * NH + j] = (_Float16)h;
    }
}

// ---------------------------------------------------------------------------
// Kernel 4: pack w2 (512x512 f32 row-major) into the per-lane WMMA B-fragment
// layout: for k-tile kt (32 K), n-tile nt (16 N):
//   lane = n%16 + 16*(k%32 >= 16), element e = k%16, 16 contiguous halves/lane
// For fixed kt the 32 n-tiles are contiguous -> any (kt, 4-ntile) slice is one
// contiguous 4 KB block, which is what the async LDS staging in k5 exploits.
// ---------------------------------------------------------------------------
__global__ void __launch_bounds__(256)
k4_pack_w2(const float* __restrict__ w2, _Float16* __restrict__ w2p) {
    int idx = blockIdx.x * 256 + threadIdx.x;      // 0 .. 512*512-1
    int k = idx >> 9, n = idx & (NH - 1);
    int kt = k >> 5, kr = k & 31, nt = n >> 4, nc = n & 15;
    int lane = nc + ((kr >= 16) ? 16 : 0);
    int e = kr & 15;
    w2p[((((kt << 5) + nt) << 5) + lane) * 16 + e] = (_Float16)w2[idx];
}

// ---------------------------------------------------------------------------
// Kernel 5: GEMM2 via v_wmma_f32_16x16x32_f16, async-LDS B staging.
// Block = 256 threads = 8 waves, covering 256 rows (2 M-tiles per wave).
// All waves share the same 64 columns, so the B tile (a contiguous 4 KB slice
// of packed w2p per K-step) is staged ONCE per block into LDS with
// global_load_async_to_lds_b128 (ASYNCcnt), double-buffered, and every wave
// reads fragments with ds_load_b128.  Each B fragment feeds 2 WMMAs.
// Epilogue: +b2, store hpre2 (f32), per-column sum/sumsq -> LDS -> global
// atomics (BN2 batch statistics).
// ---------------------------------------------------------------------------
__global__ void __launch_bounds__(256)
k5_gemm2(const _Float16* __restrict__ h1, const _Float16* __restrict__ w2p,
         const float* __restrict__ b2, float* __restrict__ hpre2,
         float* __restrict__ sum2, float* __restrict__ sumsq2) {
    __shared__ float s_sum[NH];
    __shared__ float s_sq[NH];
    __shared__ _Float16 sB[2][2048];     // 2 x 4KB B-tile staging buffers

    for (int j = threadIdx.x; j < NH; j += 256) { s_sum[j] = 0.f; s_sq[j] = 0.f; }

    const int tid  = threadIdx.x;
    const int wave = tid >> 5;
    const int lane = tid & 31;
    const int halfSel = lane >> 4;       // A: K-half select; C: M-half select
    const int mrow = lane & 15;
    const size_t m0 = (size_t)blockIdx.x * 256 + wave * 32;

    const _Float16* arow0 = h1 + (m0 + mrow) * NH;   // M-tile 0
    const _Float16* arow1 = arow0 + (size_t)16 * NH; // M-tile 1
    const int ka0 = halfSel * 16;        // lane<16: K+0/K+8 ; lane>=16: K+16/K+24
    const int ka1 = halfSel * 16 + 8;

    const unsigned ldsB0 = (unsigned)(uintptr_t)&sB[0][0];
    const unsigned ldsB1 = (unsigned)(uintptr_t)&sB[1][0];

    __syncthreads();

    for (int ng = 0; ng < 8; ++ng) {     // 8 N-groups of 64 columns
        v8f acc0[4] = {}, acc1[4] = {};
        const int n0 = ng * 64;

        // stage B tile (ng, kt): 4 KB contiguous, 256 threads x 16 B each
        auto stage = [&](int buf, int kt) {
            const _Float16* src = w2p + ((size_t)(kt * 32 + ng * 4) << 9) + tid * 8;
            unsigned ldsAddr = (buf ? ldsB1 : ldsB0) + (unsigned)tid * 16u;
            asm volatile("global_load_async_to_lds_b128 %0, %1, off"
                         :: "v"(ldsAddr), "v"((const void*)src) : "memory");
        };

        stage(0, 0);
        for (int kt = 0; kt < 16; ++kt) {
            if (kt + 1 < 16) {
                stage((kt + 1) & 1, kt + 1);
                asm volatile("s_wait_asynccnt 0x1" ::: "memory");  // stage(kt) done
            } else {
                asm volatile("s_wait_asynccnt 0x0" ::: "memory");
            }
            __syncthreads();             // all waves' slices staged

            const int k0 = kt * 32;
            union { v16h v; v8h h[2]; } A0, A1;
            A0.h[0] = *(const v8h*)(arow0 + k0 + ka0);
            A0.h[1] = *(const v8h*)(arow0 + k0 + ka1);
            A1.h[0] = *(const v8h*)(arow1 + k0 + ka0);
            A1.h[1] = *(const v8h*)(arow1 + k0 + ka1);

            const _Float16* bbuf = &sB[kt & 1][0];
            #pragma unroll
            for (int nt = 0; nt < 4; ++nt) {
                const v16h Bf = *(const v16h*)(bbuf + (((nt << 5) + lane) << 4));
                acc0[nt] = __builtin_amdgcn_wmma_f32_16x16x32_f16(
                    false, A0.v, false, Bf, (short)0, acc0[nt], false, false);
                acc1[nt] = __builtin_amdgcn_wmma_f32_16x16x32_f16(
                    false, A1.v, false, Bf, (short)0, acc1[nt], false, false);
            }
            __syncthreads();             // buffer free before reuse 2 steps later
        }

        // epilogue: C layout -> VGPR r: row = mbase + 8*halfSel + r,
        //                       col = n0 + nt*16 + mrow
        auto epilog = [&](const v8f* acc, size_t mbase) {
            #pragma unroll
            for (int nt = 0; nt < 4; ++nt) {
                const int col = n0 + nt * 16 + mrow;
                const float bb = b2[col];
                float s = 0.f, q = 0.f;
                float* dst = hpre2 + (mbase + (size_t)halfSel * 8) * NH + col;
                #pragma unroll
                for (int r = 0; r < 8; ++r) {
                    float v = acc[nt][r] + bb;
                    dst[(size_t)r * NH] = v;
                    s += v; q += v * v;
                }
                atomicAdd(&s_sum[col], s);   // lane & lane+16 share col (rows 0-7/8-15)
                atomicAdd(&s_sq[col],  q);
            }
        };
        epilog(acc0, m0);
        epilog(acc1, m0 + 16);
    }

    __syncthreads();
    for (int j = threadIdx.x; j < NH; j += 256) {
        atomicAdd(&sum2[j],   s_sum[j]);
        atomicAdd(&sumsq2[j], s_sq[j]);
    }
}

// ---------------------------------------------------------------------------
// Kernel 6: finalize BN2 statistics
// ---------------------------------------------------------------------------
__global__ void k6_bn2_stats(const float* __restrict__ sum2,
                             const float* __restrict__ sumsq2,
                             float* __restrict__ mu2, float* __restrict__ inv2,
                             float invB) {
    int j = threadIdx.x;  // one block of 512
    float mu = sum2[j] * invB;
    float var = sumsq2[j] * invB - mu * mu;
    mu2[j]  = mu;
    inv2[j] = rsqrtf(var + BN_EPS);
}

// ---------------------------------------------------------------------------
// Kernel 7: out = tanh(BN2(hpre2)) @ w3 + b3, fused streaming pass.
// One wave per row (wave32); per-row reduction with lane shuffles.
// ---------------------------------------------------------------------------
__global__ void __launch_bounds__(256)
k7_out(const float* __restrict__ hpre2, const float* __restrict__ g2,
       const float* __restrict__ beta2, const float* __restrict__ w3,
       const float* __restrict__ b3, const float* __restrict__ mu2,
       const float* __restrict__ inv2, float* __restrict__ out) {
    __shared__ float sg[NH], sb[NH], sw[NH], sm[NH], si[NH];
    for (int j = threadIdx.x; j < NH; j += 256) {
        sg[j] = g2[j]; sb[j] = beta2[j]; sw[j] = w3[j];
        sm[j] = mu2[j]; si[j] = inv2[j];
    }
    __syncthreads();
    const int wave = threadIdx.x >> 5, lane = threadIdx.x & 31;
    const size_t row = (size_t)blockIdx.x * 8 + wave;
    const float* hp = hpre2 + row * NH;
    float acc = 0.f;
    #pragma unroll 4
    for (int t = 0; t < 16; ++t) {
        int j = t * 32 + lane;
        float h = tanhf(sg[j] * (hp[j] - sm[j]) * si[j] + sb[j]);
        acc += h * sw[j];
    }
    for (int o = 16; o > 0; o >>= 1) acc += __shfl_xor(acc, o, 32);
    if (lane == 0) out[row] = acc + b3[0];
}

// ---------------------------------------------------------------------------
// Host-side launcher
// ---------------------------------------------------------------------------
extern "C" void kernel_launch(void* const* d_in, const int* in_sizes, int n_in,
                              void* d_out, int out_size, void* d_ws, size_t ws_size,
                              hipStream_t stream) {
    const float* x        = (const float*)d_in[0];
    const float* temp_set = (const float*)d_in[1];
    const float* temp_envr= (const float*)d_in[2];
    const float* cab_tab  = (const float*)d_in[3];
    const float* w1       = (const float*)d_in[4];
    const float* b1       = (const float*)d_in[5];
    const float* g1       = (const float*)d_in[6];
    const float* beta1    = (const float*)d_in[7];
    const float* w2       = (const float*)d_in[8];
    const float* b2       = (const float*)d_in[9];
    const float* g2       = (const float*)d_in[10];
    const float* beta2    = (const float*)d_in[11];
    const float* w3       = (const float*)d_in[12];
    const float* b3       = (const float*)d_in[13];
    float* out = (float*)d_out;

    const int B = in_sizes[0] / 15;          // 262144
    const float invB = 1.0f / (float)B;

    // carve workspace (256B-aligned chunks)
    char* p = (char*)d_ws;
    auto carve = [&](size_t bytes) { char* r = p; p += (bytes + 255) & ~(size_t)255; return r; };
    double*   moments = (double*)  carve(16 * sizeof(double));
    float*    mu1     = (float*)   carve(NH * sizeof(float));
    float*    inv1    = (float*)   carve(NH * sizeof(float));
    float*    sum2    = (float*)   carve(NH * sizeof(float));
    float*    sumsq2  = (float*)   carve(NH * sizeof(float));
    float*    mu2     = (float*)   carve(NH * sizeof(float));
    float*    inv2    = (float*)   carve(NH * sizeof(float));
    float*    u4      = (float*)   carve((size_t)B * NF * sizeof(float));
    _Float16* h1      = (_Float16*)carve((size_t)B * NH * sizeof(_Float16));
    _Float16* w2p     = (_Float16*)carve((size_t)NH * NH * sizeof(_Float16));
    float*    hpre2   = (float*)   carve((size_t)B * NH * sizeof(float));

    k0_zero<<<1, 256, 0, stream>>>(moments, sum2, sumsq2);
    k1_features<<<(B + 255) / 256, 256, 0, stream>>>(x, temp_set, temp_envr,
                                                     cab_tab, u4, moments, B);
    k2_bn1_stats<<<1, NH, 0, stream>>>(moments, w1, b1, mu1, inv1, invB);
    k3_h1<<<B, 256, 0, stream>>>(u4, w1, b1, g1, beta1, mu1, inv1, h1);
    k4_pack_w2<<<(NH * NH) / 256, 256, 0, stream>>>(w2, w2p);
    k5_gemm2<<<B / 256, 256, 0, stream>>>(h1, w2p, b2, hpre2, sum2, sumsq2);
    k6_bn2_stats<<<1, NH, 0, stream>>>(sum2, sumsq2, mu2, inv2, invB);
    k7_out<<<B / 8, 256, 0, stream>>>(hpre2, g2, beta2, w3, b3, mu2, inv2, out);
}
